// MultiHeadAttention_53420803228254
// MI455X (gfx1250) — matrix operand order repaired and checked by reading, not compiled, not run on hardware
//
#include <hip/hip_runtime.h>
#include <hip/hip_bf16.h>

// MHA block for MI455X (gfx1250): bf16 WMMA GEMMs + fp32 softmax/LN.
// Wave32; each wave owns a 16x16 tile; K-loop of v_wmma_f32_16x16x32_bf16.
// All GEMM operands pre-converted to bf16 with layouts chosen so every
// fragment is a contiguous 16/32-byte vector load.

typedef __attribute__((ext_vector_type(16))) __bf16 v16bf;
typedef __attribute__((ext_vector_type(8)))  __bf16 v8bf;
typedef __attribute__((ext_vector_type(8)))  float  v8f;

#define B_   8
#define S_   1024
#define D_   256
#define H_   4
#define HD_  1024
#define LN_EPS 1e-5f

__device__ __forceinline__ v8f wmma_bf16(v16bf a, v16bf b, v8f c) {
  // (neg_a, A, neg_b, B, c_mod, C, reuse_a, reuse_b)
  return __builtin_amdgcn_wmma_f32_16x16x32_bf16(false, a, false, b, (short)0, c, false, false);
}

// A-fragment (16x32, 16-bit): lane L holds row M=L%16; half=L/16 selects K-groups.
// a[j] = A[row, kbase+16*half+j] (j<16).
__device__ __forceinline__ v16bf load_a_bf16(const __bf16* A, int lda, int row, int kbase, int lane) {
  const int half = (lane >> 4) & 1;
  const __bf16* p0 = A + row * lda + kbase + 16 * half;   // 16B aligned
  v8bf lo = *(const v8bf*)p0;
  v8bf hi = *(const v8bf*)(p0 + 8);
  return __builtin_shufflevector(lo, hi, 0,1,2,3,4,5,6,7,8,9,10,11,12,13,14,15);
}

// B-fragment (32x16, 16-bit): lane L holds column N=L%16, K range (L/16)*16..+15.
// BT is N-major bf16 (K stride 1), so this is one 32B vector load.
__device__ __forceinline__ v16bf load_b_t(const __bf16* BT, int ldk, int col, int kbase, int lane) {
  const __bf16* p = BT + col * ldk + kbase + ((lane >> 4) & 1) * 16;  // 32B aligned
  return *(const v16bf*)p;
}

// ---------------------------------------------------------------------------
// Kernel 0a: weights -> bf16, transposed to N-major.  262144 threads.
// ---------------------------------------------------------------------------
__global__ __launch_bounds__(256) void prep_weights_kernel(
    const float* __restrict__ Wq, const float* __restrict__ Wk,
    const float* __restrict__ Wv, const float* __restrict__ Wp,
    __bf16* __restrict__ WqT, __bf16* __restrict__ WkT,
    __bf16* __restrict__ WvT, __bf16* __restrict__ WpT) {
  const int idx = blockIdx.x * 256 + threadIdx.x;        // [0, 262144)
  // W{q,k,v}: [256 x 1024] -> T[1024 x 256]; out index n*256+k == idx.
  const int n = idx >> 8, k = idx & 255;
  WqT[idx] = (__bf16)Wq[k * HD_ + n];
  WkT[idx] = (__bf16)Wk[k * HD_ + n];
  WvT[idx] = (__bf16)Wv[k * HD_ + n];
  // Wp: [1024 x 256] -> T[256 x 1024]; out index n2*1024+k2 == idx.
  const int n2 = idx >> 10, k2 = idx & 1023;
  WpT[idx] = (__bf16)Wp[k2 * D_ + n2];
}

// ---------------------------------------------------------------------------
// Kernel 0b: activations -> bf16 (vectorized 8-wide).  2.1M elems each.
// ---------------------------------------------------------------------------
__global__ __launch_bounds__(256) void prep_act_kernel(
    const float* __restrict__ dec, const float* __restrict__ mem,
    __bf16* __restrict__ decB, __bf16* __restrict__ memB) {
  const int idx = blockIdx.x * 256 + threadIdx.x;        // [0, 262144)
  const int o = idx * 8;
  v8f d = *(const v8f*)(dec + o);
  v8f m = *(const v8f*)(mem + o);
  v8bf db, mb;
#pragma unroll
  for (int j = 0; j < 8; ++j) { db[j] = (__bf16)d[j]; mb[j] = (__bf16)m[j]; }
  *(v8bf*)(decB + o) = db;
  *(v8bf*)(memB + o) = mb;
}

// ---------------------------------------------------------------------------
// Kernel 1: fused QKV projections.  M=B*S=8192, N=H*D=1024, K=D=256.
// Q/K stored row-major bf16 [B,H,S,D] (Q pre-scaled); V stored TRANSPOSED
// [B,H,D,S] so the attn·V GEMM gets contiguous B-fragments.
// ---------------------------------------------------------------------------
__global__ __launch_bounds__(256) void qkv_kernel(
    const __bf16* __restrict__ decB, const __bf16* __restrict__ memB,
    const __bf16* __restrict__ WqT, const float* __restrict__ bq,
    const __bf16* __restrict__ WkT, const float* __restrict__ bk,
    const __bf16* __restrict__ WvT, const float* __restrict__ bv,
    __bf16* __restrict__ Qb, __bf16* __restrict__ Kb, __bf16* __restrict__ Vt) {
  const int lane = threadIdx.x & 31;
  const int wave = threadIdx.x >> 5;
  const int tile = blockIdx.x * 8 + wave;        // 32768 tiles
  const int mt = tile >> 6;                      // 512 m-tiles
  const int nt = tile & 63;                      // 64 n-tiles
  const int half = lane >> 4;
  const int nl = lane & 15;
  const int arow = mt * 16 + nl;
  const int col = nt * 16 + nl;

  v8f aq = {}, ak = {}, av = {};
#pragma unroll
  for (int kk = 0; kk < D_; kk += 32) {
    v16bf aD = load_a_bf16(decB, D_, arow, kk, lane);
    v16bf aM = load_a_bf16(memB, D_, arow, kk, lane);
    v16bf bQ = load_b_t(WqT, D_, col, kk, lane);
    v16bf bK = load_b_t(WkT, D_, col, kk, lane);
    v16bf bV = load_b_t(WvT, D_, col, kk, lane);
    aq = wmma_bf16(aD, bQ, aq);
    ak = wmma_bf16(aM, bK, ak);
    av = wmma_bf16(aM, bV, av);
  }
  const float scale = 0.0625f;                   // 1/sqrt(256)
  const int h = col >> 8;
  const int d = col & 255;
  const int m0 = mt * 16 + 8 * half;             // first row of this column
  const int b = m0 >> 10;
  const int s0 = m0 & 1023;                      // tile never crosses batch
  const int bh = b * H_ + h;
  v8bf vpack;
#pragma unroll
  for (int r = 0; r < 8; ++r) {
    const int o = (bh * S_ + s0 + r) * D_ + d;
    Qb[o] = (__bf16)((aq[r] + bq[col]) * scale);
    Kb[o] = (__bf16)(ak[r] + bk[col]);
    vpack[r] = (__bf16)(av[r] + bv[col]);
  }
  *(v8bf*)(Vt + (bh * D_ + d) * S_ + s0) = vpack;  // contiguous in s
}

// ---------------------------------------------------------------------------
// Kernel 2: scores = Qs·K^T, mask, softmax, *query_mask -> attn (fp32, d_out)
// plus bf16 shadow for the attn·V GEMM.  One block per (b,h,16-query strip).
// The int32 mask is the one HBM-bound input: prefetch it a tile ahead.
// ---------------------------------------------------------------------------
__global__ __launch_bounds__(256) void attn_softmax_kernel(
    const __bf16* __restrict__ Qb, const __bf16* __restrict__ Kb,
    const int* __restrict__ mask, const int* __restrict__ qmask,
    float* __restrict__ attnOut, __bf16* __restrict__ attnB) {
  __shared__ float sc[16][S_];                   // 64 KB
  const int lane = threadIdx.x & 31;
  const int wave = threadIdx.x >> 5;
  const int qt = blockIdx.x & 63;
  const int bh = blockIdx.x >> 6;                // b*H + h
  const int b = bh >> 2;
  const int q0 = qt * 16;
  const int half = lane >> 4;
  const int nl = lane & 15;
  const __bf16* Qrows = Qb + (bh * S_ + q0) * D_;
  const __bf16* Kbase = Kb + bh * S_ * D_;

  // Q strip fragments are invariant across key tiles: load once (64 VGPRs).
  v16bf qa[8];
#pragma unroll
  for (int i = 0; i < 8; ++i) qa[i] = load_a_bf16(Qrows, D_, nl, i * 32, lane);

  // Phase 1: each wave computes 8 of the 64 key-tiles for this query strip.
#pragma unroll 1
  for (int kt = wave; kt < 64; kt += 8) {
    const int key = kt * 16 + nl;                // this lane's key column
    const __bf16* kp = Kbase + key * D_ + half * 16;
    v8f acc = {};
#pragma unroll
    for (int i = 0; i < 8; ++i)
      acc = wmma_bf16(qa[i], *(const v16bf*)(kp + i * 32), acc);
#pragma unroll
    for (int r = 0; r < 8; ++r) {
      const int qr = r + 8 * half;
      const int kc = kt * 16 + nl;
      const int midx = (b * S_ + q0 + qr) * S_ + kc;
      const int mv = mask[midx];
      if (kt + 8 < 64) __builtin_prefetch(&mask[midx + 128], 0, 1);  // next tile
      sc[qr][kc] = (mv == 0) ? -1e9f : acc[r];
    }
  }
  __syncthreads();

  // Phase 2: wave w does rows 2w, 2w+1; wave32 shfl reductions.
#pragma unroll 1
  for (int rr = 0; rr < 2; ++rr) {
    const int qr = wave * 2 + rr;
    float* srow = sc[qr];
    float mx = -3.4e38f;
#pragma unroll 1
    for (int j = lane; j < S_; j += 32) mx = fmaxf(mx, srow[j]);
#pragma unroll
    for (int off = 16; off > 0; off >>= 1) mx = fmaxf(mx, __shfl_xor(mx, off, 32));
    float sum = 0.f;
#pragma unroll 1
    for (int j = lane; j < S_; j += 32) {
      float e = __expf(srow[j] - mx);
      srow[j] = e;
      sum += e;
    }
#pragma unroll
    for (int off = 16; off > 0; off >>= 1) sum += __shfl_xor(sum, off, 32);
    const float f = (float)qmask[b * S_ + q0 + qr] / sum;
    const int rowoff = (bh * S_ + q0 + qr) * S_;
    float*  arow  = attnOut + rowoff;
    __bf16* abrow = attnB + rowoff;
#pragma unroll 1
    for (int j = lane; j < S_; j += 32) {
      const float v = srow[j] * f;
      arow[j]  = v;
      abrow[j] = (__bf16)v;
    }
  }
}

// ---------------------------------------------------------------------------
// Kernel 3: Z = attn·V per (b,h); store transposed into head-mixed buffer
// X2[b, h*256+d, q] (bf16) — implements transpose(0,1,3,2)+reshape directly.
// ---------------------------------------------------------------------------
__global__ __launch_bounds__(256) void attn_v_kernel(
    const __bf16* __restrict__ attnB, const __bf16* __restrict__ Vt,
    __bf16* __restrict__ X2) {
  const int lane = threadIdx.x & 31;
  const int wave = threadIdx.x >> 5;
  const int tile = blockIdx.x * 8 + wave;        // 32768 tiles
  const int bh = tile >> 10;
  const int t = tile & 1023;
  const int qt = t >> 4;                         // 64 q-tiles
  const int dt = t & 15;                         // 16 d-tiles
  const int half = lane >> 4;
  const int nl = lane & 15;
  const __bf16* Arows = attnB + (bh * S_ + qt * 16) * S_;
  const int d = dt * 16 + nl;
  const __bf16* vp = Vt + (bh * D_ + d) * S_ + half * 16;  // contiguous in s

  v8f acc = {};
#pragma unroll 4
  for (int kk = 0; kk < S_; kk += 32) {
    v16bf a = load_a_bf16(Arows, S_, nl, kk, lane);
    acc = wmma_bf16(a, *(const v16bf*)(vp + kk), acc);
  }
  const int b = bh >> 2;
  const int h = bh & 3;
  const int q0 = qt * 16 + 8 * half;
  v8bf zpack;
#pragma unroll
  for (int r = 0; r < 8; ++r) zpack[r] = (__bf16)acc[r];
  *(v8bf*)(X2 + (b * S_ + h * D_ + d) * S_ + q0) = zpack;  // contiguous in q
}

// ---------------------------------------------------------------------------
// Kernel 4: out = LN(X2·WpT + bp + residual).  One block per 16-row strip.
// ---------------------------------------------------------------------------
__global__ __launch_bounds__(256) void proj_ln_kernel(
    const __bf16* __restrict__ X2, const __bf16* __restrict__ WpT,
    const float* __restrict__ bp, const float* __restrict__ dec,
    const float* __restrict__ gamma, const float* __restrict__ beta,
    float* __restrict__ out) {
  __shared__ float sx[16][D_];                   // 16 KB
  const int lane = threadIdx.x & 31;
  const int wave = threadIdx.x >> 5;
  const int half = lane >> 4;
  const int nl = lane & 15;
  const int m0 = blockIdx.x * 16;
  const __bf16* Arows = X2 + m0 * HD_;

#pragma unroll 1
  for (int t = 0; t < 2; ++t) {
    const int col = (wave * 2 + t) * 16 + nl;    // 16 n-tiles / 8 waves
    v8f acc = {};
#pragma unroll 4
    for (int kk = 0; kk < HD_; kk += 32) {
      v16bf a  = load_a_bf16(Arows, HD_, nl, kk, lane);
      v16bf bf = load_b_t(WpT, HD_, col, kk, lane);
      acc = wmma_bf16(a, bf, acc);
    }
#pragma unroll
    for (int r = 0; r < 8; ++r) {
      const int rl = r + 8 * half;
      const int m = m0 + rl;
      sx[rl][col] = acc[r] + bp[col] + dec[m * D_ + col];
    }
  }
  __syncthreads();

#pragma unroll 1
  for (int rr = 0; rr < 2; ++rr) {
    const int rl = wave * 2 + rr;
    const int m = m0 + rl;
    const float* row = sx[rl];
    float s = 0.f, s2 = 0.f;
#pragma unroll
    for (int j = lane; j < D_; j += 32) { float v = row[j]; s += v; s2 += v * v; }
#pragma unroll
    for (int off = 16; off > 0; off >>= 1) {
      s  += __shfl_xor(s, off, 32);
      s2 += __shfl_xor(s2, off, 32);
    }
    const float mu = s * (1.0f / D_);
    const float var = s2 * (1.0f / D_) - mu * mu;
    const float rstd = rsqrtf(var + LN_EPS);
#pragma unroll
    for (int j = lane; j < D_; j += 32)
      out[m * D_ + j] = (row[j] - mu) * rstd * gamma[j] + beta[j];
  }
}

// ---------------------------------------------------------------------------
extern "C" void kernel_launch(void* const* d_in, const int* in_sizes, int n_in,
                              void* d_out, int out_size, void* d_ws, size_t ws_size,
                              hipStream_t stream) {
  const float* mem   = (const float*)d_in[0];
  const float* dec   = (const float*)d_in[1];
  const int*   mask  = (const int*)d_in[2];
  const int*   qmask = (const int*)d_in[3];
  const float* Wq = (const float*)d_in[4];
  const float* Wk = (const float*)d_in[6];
  const float* Wv = (const float*)d_in[8];
  const float* Wp = (const float*)d_in[10];
  const float* bq = (const float*)d_in[5];
  const float* bk = (const float*)d_in[7];
  const float* bv = (const float*)d_in[9];
  const float* bp = (const float*)d_in[11];
  const float* gamma = (const float*)d_in[12];
  const float* beta  = (const float*)d_in[13];

  float* out  = (float*)d_out;
  float* attn = out + (size_t)B_ * S_ * D_;      // second tuple output

  const size_t qkvElems  = (size_t)B_ * H_ * S_ * D_;   // 8.39M
  const size_t attnElems = (size_t)B_ * H_ * S_ * S_;   // 33.6M
  const size_t wElems    = (size_t)D_ * HD_;            // 262144
  const size_t actElems  = (size_t)B_ * S_ * D_;        // 2.1M
  __bf16* Qb    = (__bf16*)d_ws;
  __bf16* Kb    = Qb + qkvElems;
  __bf16* Vt    = Kb + qkvElems;                 // [B,H,D,S]
  __bf16* X2    = Vt + qkvElems;                 // [B, H*D(mixed s'), S(q)]
  __bf16* attnB = X2 + qkvElems;                 // bf16 shadow of attn
  __bf16* WqT   = attnB + attnElems;
  __bf16* WkT   = WqT + wElems;
  __bf16* WvT   = WkT + wElems;
  __bf16* WpT   = WvT + wElems;
  __bf16* decB  = WpT + wElems;                  // bf16 activations
  __bf16* memB  = decB + actElems;

  prep_weights_kernel<<<1024, 256, 0, stream>>>(Wq, Wk, Wv, Wp, WqT, WkT, WvT, WpT);
  prep_act_kernel<<<1024, 256, 0, stream>>>(dec, mem, decB, memB);
  qkv_kernel<<<4096, 256, 0, stream>>>(decB, memB, WqT, bq, WkT, bk, WvT, bv, Qb, Kb, Vt);
  attn_softmax_kernel<<<2048, 256, 0, stream>>>(Qb, Kb, mask, qmask, attn, attnB);
  attn_v_kernel<<<4096, 256, 0, stream>>>(attnB, Vt, X2);
  proj_ln_kernel<<<512, 256, 0, stream>>>(X2, WpT, bp, dec, gamma, beta, out);
}
